// RNModule_47983374631060
// MI455X (gfx1250) — matrix operand order repaired and checked by reading, not compiled, not run on hardware
//
#include <hip/hip_runtime.h>
#include <hip/hip_bf16.h>

// ---------------------------------------------------------------------------
// RNModule fused forward for MI455X (gfx1250, wave32, WMMA f16->f32).
// BS=16, N=1024, PAIR=8, C=128.  Outputs: rn_feature [16,128,1024] f32
// followed by logits_0 [16,2,8192] f32 (concatenated flat in d_out).
//
// d_in order assumption (setup_inputs dict insertion order, params nested
// dict flattened in insertion order):
//  0 feature | 1 W_l1 2 b_l1 3 bn1_g 4 bn1_b 5 bn1_m 6 bn1_v
//  7 W_l2 8 b_l2 9 bn2_g 10 bn2_b 11 bn2_m 12 bn2_v
// 13 W_q 14 b_q 15 W_k 16 b_k 17 W_rn1 18 b_rn1
// 19 W_rn2 20 b_rn2 21 rnbn1_g 22 rnbn1_b 23 rnbn1_m 24 rnbn1_v
// 25 W_rn3 26 b_rn3 27 rnbn2_g 28 rnbn2_b 29 rnbn2_m 30 rnbn2_v
// 31 W_rn4 32 b_rn4 | 33 idx_obj 34 idx_all
// ---------------------------------------------------------------------------

typedef _Float16 v16h __attribute__((ext_vector_type(16)));
typedef float    v8f  __attribute__((ext_vector_type(8)));

#define BS   16
#define NPT  1024
#define PAIR 8
#define C    128

// per-workgroup tile: 8 points -> 64 pair-rows
#define NB    8
#define ROWS  64
#define THREADS 256
#define WAVES 8

// LDS row strides (halfs), padded by 8 halfs (16B) to stagger banks while
// keeping 16B alignment for b128 LDS loads.
#define STR1 264   // 256-wide activation buffers
#define STR2 136   // 128-wide activation buffer

// LDS carve (bytes)
#define OFF_RELU 0                    // _Float16 [64][STR1]  (rel_u, later ra2)
#define OFF_H    33792                // _Float16 [64][STR1]  (h, later ra)
#define OFF_RL   67584                // _Float16 [64][STR2]  (rel_local)
#define OFF_XQ   84992                // float    [64][32]
#define OFF_XK   93184                // float    [8][32]
#define OFF_ATT  94208                // float    [64]
#define OFF_AGG  94464                // _Float16 [8][128]
#define OFF_J    96512                // int      [64]
#define SMEM_BYTES 96768

// ---------------- WMMA helpers (CDNA5 layouts) -----------------------------
// A 16x32 f16: lane&15 = row M; lane>>4 selects K-half.
//   elements 0..7 -> K = kh*8 + i ; elements 8..15 -> K = kh*8 + 16 + i
// row: caller-resolved LDS row index for this lane.
static __device__ inline v16h load_a_tile(const _Float16* lds, int stride,
                                          int row, int k0, int lane) {
  const int kh = ((lane >> 4) & 1) * 8;
  const _Float16* p = lds + row * stride + k0 + kh;
  v16h a;
#pragma unroll
  for (int i = 0; i < 8; ++i) { a[i] = p[i]; a[i + 8] = p[i + 16]; }
  return a;
}

// B 32x16 f16 from a row-major f16 weight W[O][K]:
//   lane&15 = output column o; lane>>4 selects K-half (same K order as A).
static __device__ inline v16h load_b_tile(const _Float16* W, int ldw,
                                          int o0, int k0, int lane) {
  const int o  = o0 + (lane & 15);
  const int kb = k0 + ((lane >> 4) & 1) * 8;
  const _Float16* p = W + o * ldw + kb;
  v16h b;
#pragma unroll
  for (int i = 0; i < 8; ++i) { b[i] = p[i]; b[i + 8] = p[i + 16]; }
  return b;
}

// Full K-loop 16x16 tile:  A rows from LDS, B from global f16 weights.
static __device__ inline v8f gemm_tile(const _Float16* A, int lda, int rowRes,
                                       const _Float16* W, int ldw, int o0,
                                       int K, int lane) {
  v8f acc = {0.f, 0.f, 0.f, 0.f, 0.f, 0.f, 0.f, 0.f};
  for (int k0 = 0; k0 < K; k0 += 32) {
    v16h a = load_a_tile(A, lda, rowRes, k0, lane);
    v16h b = load_b_tile(W, ldw, o0, k0, lane);
    acc = __builtin_amdgcn_wmma_f32_16x16x32_f16(
        false, a, false, b, (short)0, acc, false, false);
  }
  return acc;
}

// D 16x16 f32: VGPR j -> row j (lanes 0-15) / j+8 (lanes 16-31); col = lane&15.
// Epilogue: y = relu(x*scale[col] + shift[col]), stored as f16 in LDS.
static __device__ inline void store_bn_relu(_Float16* lds, int stride, int r0,
                                            int colBase, v8f c,
                                            const float* scale,
                                            const float* shift, int lane) {
  const int col = colBase + (lane & 15);
  const float sc = scale[col], sh = shift[col];
  const int mh = ((lane >> 4) & 1) * 8;
#pragma unroll
  for (int j = 0; j < 8; ++j) {
    float v = c[j] * sc + sh;
    lds[(r0 + mh + j) * stride + col] = (_Float16)(v > 0.f ? v : 0.f);
  }
}

// ---------------- prep kernels ---------------------------------------------
// feature [B,C,N] f32 -> featT [B,N,C] f16 (coalesced f16 writes; strided
// reads are L2-resident: whole tensor is 8MB << 192MB L2).
__global__ void k_transpose(const float* __restrict__ feature,
                            _Float16* __restrict__ featT) {
  int idx = blockIdx.x * blockDim.x + threadIdx.x;   // [b][n][c], c fastest
  int b = idx >> 17;
  int n = (idx >> 7) & (NPT - 1);
  int c = idx & (C - 1);
  featT[idx] = (_Float16)feature[((b * C + c) << 10) + n];
}

// convert all GEMM weights to f16, concatenated in ws in this order
__global__ void k_weights(const float* __restrict__ w0,  // W_l1  65536
                          const float* __restrict__ w1,  // W_l2  32768
                          const float* __restrict__ w2,  // W_q    4096
                          const float* __restrict__ w3,  // W_k    4096
                          const float* __restrict__ w4,  // W_rn1 16384
                          const float* __restrict__ w5,  // W_rn2 16384
                          const float* __restrict__ w6,  // W_rn3 16384
                          _Float16* __restrict__ out) {
  int i = blockIdx.x * blockDim.x + threadIdx.x;
  if (i >= 155648) return;
  float v;
  if      (i <  65536) v = w0[i];
  else if (i <  98304) v = w1[i -  65536];
  else if (i < 102400) v = w2[i -  98304];
  else if (i < 106496) v = w3[i - 102400];
  else if (i < 122880) v = w4[i - 106496];
  else if (i < 139264) v = w5[i - 122880];
  else                 v = w6[i - 139264];
  out[i] = (_Float16)v;
}

// fold conv-bias + inference BN:  y = x*s + h,  s=g*rsqrt(v+eps),
// h = bn_b - bn_m*s + bias*s.   f layout: s1[256] h1[256] s2[128] h2[128]
// srn2[128] hrn2[128] srn3[128] hrn3[128]
__global__ void k_fuse(const float* bl1, const float* g1, const float* bb1,
                       const float* m1, const float* v1,
                       const float* bl2, const float* g2, const float* bb2,
                       const float* m2, const float* v2,
                       const float* brn2, const float* g3, const float* bb3,
                       const float* m3, const float* v3,
                       const float* brn3, const float* g4, const float* bb4,
                       const float* m4, const float* v4,
                       float* __restrict__ f) {
  int i = threadIdx.x;
  if (i < 256) {
    float s = g1[i] * rsqrtf(v1[i] + 1e-5f);
    f[i] = s; f[256 + i] = bb1[i] - m1[i] * s + bl1[i] * s;
  }
  if (i < 128) {
    float s = g2[i] * rsqrtf(v2[i] + 1e-5f);
    f[512 + i] = s; f[640 + i] = bb2[i] - m2[i] * s + bl2[i] * s;
    s = g3[i] * rsqrtf(v3[i] + 1e-5f);
    f[768 + i] = s; f[896 + i] = bb3[i] - m3[i] * s + brn2[i] * s;
    s = g4[i] * rsqrtf(v4[i] + 1e-5f);
    f[1024 + i] = s; f[1152 + i] = bb4[i] - m4[i] * s + brn3[i] * s;
  }
}

// ---------------- fused main kernel ----------------------------------------
__global__ void __launch_bounds__(THREADS)
k_rn_fused(const _Float16* __restrict__ featT,
           const int* __restrict__ idx_obj, const int* __restrict__ idx_all,
           const _Float16* __restrict__ Wl1h, const _Float16* __restrict__ Wl2h,
           const _Float16* __restrict__ Wqh,  const _Float16* __restrict__ Wkh,
           const _Float16* __restrict__ Wrn1h, const _Float16* __restrict__ Wrn2h,
           const _Float16* __restrict__ Wrn3h,
           const float* __restrict__ fused,
           const float* __restrict__ b_q, const float* __restrict__ b_k,
           const float* __restrict__ b_rn1,
           const float* __restrict__ W_rn4, const float* __restrict__ b_rn4,
           float* __restrict__ out) {
  extern __shared__ char smem[];
  _Float16* sRelu = (_Float16*)(smem + OFF_RELU);  // rel_u / ra2
  _Float16* sH    = (_Float16*)(smem + OFF_H);     // h / ra
  _Float16* sRL   = (_Float16*)(smem + OFF_RL);    // rel_local
  float*    sXQ   = (float*)(smem + OFF_XQ);
  float*    sXK   = (float*)(smem + OFF_XK);
  float*    sAtt  = (float*)(smem + OFF_ATT);
  _Float16* sAgg  = (_Float16*)(smem + OFF_AGG);
  int*      sJ    = (int*)(smem + OFF_J);

  const int tid  = threadIdx.x;
  const int lane = tid & 31;
  const int wave = tid >> 5;
  const int b    = blockIdx.x >> 7;          // 128 blocks per batch
  const int n0   = (blockIdx.x & 127) * NB;  // first point of this block

  const float* s1 = fused,        *h1 = fused + 256;
  const float* s2 = fused + 512,  *h2 = fused + 640;
  const float* srn2 = fused + 768,  *hrn2 = fused + 896;
  const float* srn3 = fused + 1024, *hrn3 = fused + 1152;

  // ---- S0: neighbor index resolve + build rel_u = [feat_i | feat_i-feat_j]
  if (tid < ROWS) {
    int pt = tid >> 3, p = tid & 7;
    int ia = idx_all[(((b << 10) + n0 + pt) << 3) + p];
    sJ[tid] = idx_obj[(b << 10) + ia];
  }
  __syncthreads();
  for (int e = tid; e < ROWS * C; e += THREADS) {
    int r = e >> 7, c = e & (C - 1);
    _Float16 fi = featT[(((b << 10) + n0 + (r >> 3)) << 7) + c];
    _Float16 fj = featT[(((b << 10) + sJ[r]) << 7) + c];
    sRelu[r * STR1 + c]       = fi;
    sRelu[r * STR1 + 128 + c] = (_Float16)(fi - fj);
  }
  __syncthreads();

  // ---- S1: h = relu(bn1(rel_u @ W_l1^T))   [64x256] x [256x256]
  for (int t = wave; t < 64; t += WAVES) {
    int r0 = (t >> 4) * 16, o0 = (t & 15) * 16;
    v8f acc = gemm_tile(sRelu, STR1, r0 + (lane & 15), Wl1h, 256, o0, 256, lane);
    store_bn_relu(sH, STR1, r0, o0, acc, s1, h1, lane);
  }
  __syncthreads();

  // ---- S2: rel_local = relu(bn2(h @ W_l2^T))  [64x256] x [256x128]
  for (int t = wave; t < 32; t += WAVES) {
    int r0 = (t >> 3) * 16, o0 = (t & 7) * 16;
    v8f acc = gemm_tile(sH, STR1, r0 + (lane & 15), Wl2h, 256, o0, 256, lane);
    store_bn_relu(sRL, STR2, r0, o0, acc, s2, h2, lane);
  }
  __syncthreads();

  // ---- S3: x_q = tanh(rl @ W_q^T + b_q)  and  x_k = tanh(feat @ W_k^T + b_k)
  for (int t = wave; t < 10; t += WAVES) {
    if (t < 8) {  // x_q: 4 row-tiles x 2 col-tiles
      int r0 = (t >> 1) * 16, o0 = (t & 1) * 16;
      v8f acc = gemm_tile(sRL, STR2, r0 + (lane & 15), Wqh, 128, o0, 128, lane);
      int col = o0 + (lane & 15);
      float bq = b_q[col];
      int mh = ((lane >> 4) & 1) * 8;
#pragma unroll
      for (int j = 0; j < 8; ++j)
        sXQ[(r0 + mh + j) * 32 + col] = tanhf(acc[j] + bq);
    } else {      // x_k: 8 valid rows (fi lives in sRelu row pt*8, cols 0..127)
      int o0 = (t - 8) * 16;
      int rowRes = ((lane & 15) & 7) * 8;
      v8f acc = gemm_tile(sRelu, STR1, rowRes, Wkh, 128, o0, 128, lane);
      if ((lane >> 4) == 0) {  // rows 0..7 only
        int col = o0 + (lane & 15);
        float bk = b_k[col];
#pragma unroll
        for (int j = 0; j < 8; ++j)
          sXK[j * 32 + col] = tanhf(acc[j] + bk);
      }
    }
  }
  __syncthreads();

  // ---- S4: energy + softmax over the 8 neighbors of each point
  if (tid < ROWS) {
    float e = 0.f;
    int pt = tid >> 3;
#pragma unroll
    for (int q = 0; q < 32; ++q) e += sXQ[tid * 32 + q] * sXK[pt * 32 + q];
    sAtt[tid] = e;
  }
  __syncthreads();
  if (tid < NB) {
    float mx = -3.4e38f;
#pragma unroll
    for (int p = 0; p < PAIR; ++p) mx = fmaxf(mx, sAtt[tid * 8 + p]);
    float ex[PAIR], s = 0.f;
#pragma unroll
    for (int p = 0; p < PAIR; ++p) { ex[p] = expf(sAtt[tid * 8 + p] - mx); s += ex[p]; }
    float inv = 1.f / s;  // explicit renorm in ref is identity after softmax
#pragma unroll
    for (int p = 0; p < PAIR; ++p) sAtt[tid * 8 + p] = ex[p] * inv;
  }
  __syncthreads();

  // ---- S5: agg[pt][c] = sum_p att * rel_local
  for (int e = tid; e < NB * C; e += THREADS) {
    int pt = e >> 7, c = e & (C - 1);
    float s = 0.f;
#pragma unroll
    for (int p = 0; p < PAIR; ++p)
      s += sAtt[pt * 8 + p] * (float)sRL[(pt * 8 + p) * STR2 + c];
    sAgg[pt * C + c] = (_Float16)s;
  }
  __syncthreads();

  // ---- S6: rn_feature = agg @ W_rn1^T + b_rn1, written as [b,128,N]
  for (int t = wave; t < 8; t += WAVES) {
    int o0 = t * 16;
    int rowRes = (lane & 15) & 7;  // sAgg rows 0..7 (dup for 8..15, discarded)
    v8f acc = gemm_tile(sAgg, C, rowRes, Wrn1h, 128, o0, 128, lane);
    if ((lane >> 4) == 0) {
      int col = o0 + (lane & 15);
      float br = b_rn1[col];
#pragma unroll
      for (int j = 0; j < 8; ++j)
        out[(((b << 7) + col) << 10) + n0 + j] = acc[j] + br;
    }
  }
  __syncthreads();

  // ---- S7: ra = relu(rnbn1(rl @ W_rn2^T))  -> sH (reuse)
  for (int t = wave; t < 32; t += WAVES) {
    int r0 = (t >> 3) * 16, o0 = (t & 7) * 16;
    v8f acc = gemm_tile(sRL, STR2, r0 + (lane & 15), Wrn2h, 128, o0, 128, lane);
    store_bn_relu(sH, STR1, r0, o0, acc, srn2, hrn2, lane);
  }
  __syncthreads();

  // ---- S8: ra2 = relu(rnbn2(ra @ W_rn3^T)) -> sRelu (reuse)
  for (int t = wave; t < 32; t += WAVES) {
    int r0 = (t >> 3) * 16, o0 = (t & 7) * 16;
    v8f acc = gemm_tile(sH, STR1, r0 + (lane & 15), Wrn3h, 128, o0, 128, lane);
    store_bn_relu(sRelu, STR1, r0, o0, acc, srn3, hrn3, lane);
  }
  __syncthreads();

  // ---- S9: logits (out=2, too narrow for WMMA -> VALU dot), layout [b,2,N*P]
  if (tid < ROWS) {
    float z0 = b_rn4[0], z1 = b_rn4[1];
#pragma unroll 8
    for (int c = 0; c < C; ++c) {
      float x = (float)sRelu[tid * STR1 + c];
      z0 += x * W_rn4[c];
      z1 += x * W_rn4[128 + c];
    }
    int base = (BS * C * NPT) + ((b * 2) << 13) + (n0 << 3) + tid;
    out[base] = z0;
    out[base + 8192] = z1;
  }
}

// ---------------------------------------------------------------------------
extern "C" void kernel_launch(void* const* d_in, const int* in_sizes, int n_in,
                              void* d_out, int out_size, void* d_ws, size_t ws_size,
                              hipStream_t stream) {
  const float* feature = (const float*)d_in[0];
  const float* W_l1 = (const float*)d_in[1];
  const float* b_l1 = (const float*)d_in[2];
  const float* bn1_g = (const float*)d_in[3];
  const float* bn1_b = (const float*)d_in[4];
  const float* bn1_m = (const float*)d_in[5];
  const float* bn1_v = (const float*)d_in[6];
  const float* W_l2 = (const float*)d_in[7];
  const float* b_l2 = (const float*)d_in[8];
  const float* bn2_g = (const float*)d_in[9];
  const float* bn2_b = (const float*)d_in[10];
  const float* bn2_m = (const float*)d_in[11];
  const float* bn2_v = (const float*)d_in[12];
  const float* W_q = (const float*)d_in[13];
  const float* b_q = (const float*)d_in[14];
  const float* W_k = (const float*)d_in[15];
  const float* b_k = (const float*)d_in[16];
  const float* W_rn1 = (const float*)d_in[17];
  const float* b_rn1 = (const float*)d_in[18];
  const float* W_rn2 = (const float*)d_in[19];
  const float* b_rn2 = (const float*)d_in[20];
  const float* rnbn1_g = (const float*)d_in[21];
  const float* rnbn1_b = (const float*)d_in[22];
  const float* rnbn1_m = (const float*)d_in[23];
  const float* rnbn1_v = (const float*)d_in[24];
  const float* W_rn3 = (const float*)d_in[25];
  const float* b_rn3 = (const float*)d_in[26];
  const float* rnbn2_g = (const float*)d_in[27];
  const float* rnbn2_b = (const float*)d_in[28];
  const float* rnbn2_m = (const float*)d_in[29];
  const float* rnbn2_v = (const float*)d_in[30];
  const float* W_rn4 = (const float*)d_in[31];
  const float* b_rn4 = (const float*)d_in[32];
  const int* idx_obj = (const int*)d_in[33];
  const int* idx_all = (const int*)d_in[34];
  float* out = (float*)d_out;

  // workspace layout (halfs unless noted): featT 2097152 | Wl1 65536 |
  // Wl2 32768 | Wq 4096 | Wk 4096 | Wrn1 16384 | Wrn2 16384 | Wrn3 16384 |
  // fused f32 1280.  Total ~4.31 MB.
  _Float16* featT = (_Float16*)d_ws;
  _Float16* whBase = featT + (size_t)BS * NPT * C;
  _Float16* Wl1h  = whBase;
  _Float16* Wl2h  = Wl1h + 65536;
  _Float16* Wqh   = Wl2h + 32768;
  _Float16* Wkh   = Wqh + 4096;
  _Float16* Wrn1h = Wkh + 4096;
  _Float16* Wrn2h = Wrn1h + 16384;
  _Float16* Wrn3h = Wrn2h + 16384;
  float* fusedP   = (float*)(Wrn3h + 16384);

  k_transpose<<<(BS * NPT * C) / 256, 256, 0, stream>>>(feature, featT);
  k_weights<<<155648 / 256, 256, 0, stream>>>(W_l1, W_l2, W_q, W_k, W_rn1,
                                              W_rn2, W_rn3, whBase);
  k_fuse<<<1, 256, 0, stream>>>(b_l1, bn1_g, bn1_b, bn1_m, bn1_v,
                                b_l2, bn2_g, bn2_b, bn2_m, bn2_v,
                                b_rn2, rnbn1_g, rnbn1_b, rnbn1_m, rnbn1_v,
                                b_rn3, rnbn2_g, rnbn2_b, rnbn2_m, rnbn2_v,
                                fusedP);
  k_rn_fused<<<BS * (NPT / NB), THREADS, SMEM_BYTES, stream>>>(
      featT, idx_obj, idx_all, Wl1h, Wl2h, Wqh, Wkh, Wrn1h, Wrn2h, Wrn3h,
      fusedP, b_q, b_k, b_rn1, W_rn4, b_rn4, out);
}
